// GINConv_687194767736
// MI455X (gfx1250) — matrix operand order repaired and checked by reading, not compiled, not run on hardware
//
#include <hip/hip_runtime.h>
#include <stdint.h>

typedef __attribute__((ext_vector_type(16))) __bf16 v16bf;
typedef __attribute__((ext_vector_type(8)))  float  v8f;

#define NNODES  100000
#define DEG     32
#define CIN     128
#define COUT    128
#define TILE_M  128
#define THREADS 256
#define RS      136                         // padded LDS row stride in bf16 elems (272 B)
#define WPACK_ELEMS (8*4*32*16)             // nt * kchunk * lane * 16 halfs

union FragBF { v16bf v; uint4 q[2]; };

__global__ void __launch_bounds__(THREADS)
gin_fused_kernel(const float* __restrict__ x,
                 const long long* __restrict__ eidx,
                 const float* __restrict__ eps,
                 const float* __restrict__ W,
                 const float* __restrict__ bias,
                 float* __restrict__ out)
{
    extern __shared__ char smem[];
    __bf16* hHi = (__bf16*)smem;                    // [TILE_M][RS]
    __bf16* hLo = hHi + TILE_M * RS;
    __bf16* wHi = hLo + TILE_M * RS;                // fragment-packed W^T
    __bf16* wLo = wHi + WPACK_ELEMS;

    const int tid  = threadIdx.x;
    const int lane = tid & 31;
    const int wv   = tid >> 5;                      // wave id 0..7
    const int blockBase = blockIdx.x * TILE_M;
    const float epsv = 1.0f + eps[0];

    // ---------- Phase A: pack W^T (bf16 hi+lo) in WMMA B-fragment order ----------
    // fragment half j of lane l, chunk kkc, n-tile nt holds Wt[k][n] = W[n][k]
    // with k = kkc*32 + (l>=16 ? 8 : 0) + (j&7) + (j>=8 ? 16 : 0), n = nt*16 + (l&15)
    for (int e = tid; e < WPACK_ELEMS; e += THREADS) {
        int j   = e & 15;
        int ln  = (e >> 4) & 31;
        int kkc = (e >> 9) & 3;
        int nt  = e >> 11;
        int k = kkc * 32 + ((ln & 16) ? 8 : 0) + (j & 7) + ((j & 8) ? 16 : 0);
        int n = nt * 16 + (ln & 15);
        float v = W[n * CIN + k];
        __bf16 hi = (__bf16)v;
        __bf16 lo = (__bf16)(v - (float)hi);
        wHi[e] = hi;
        wLo[e] = lo;
    }

    // ---------- Phase B: neighbor aggregation -> h rows in LDS (bf16 hi+lo) ------
    const float4* x4 = reinterpret_cast<const float4*>(x);
    for (int r = 0; r < 16; ++r) {
        int m = wv * 16 + r;
        int g = blockBase + m;
        float4 acc = make_float4(0.f, 0.f, 0.f, 0.f);
        if (g < NNODES) {
            float4 xv = x4[g * (CIN / 4) + lane];   // lane covers 4 channels
            acc.x = epsv * xv.x; acc.y = epsv * xv.y;
            acc.z = epsv * xv.z; acc.w = epsv * xv.w;
            int myidx = (int)eidx[g * DEG + lane];  // one coalesced idx load
            #pragma unroll 4
            for (int j = 0; j < DEG; ++j) {
                int idx = __shfl(myidx, j, 32);     // wave-uniform neighbor id
                if (idx < NNODES) {                 // idx==NNODES -> zero pad row
                    float4 nv = x4[idx * (CIN / 4) + lane];  // 512B row, L2-resident
                    acc.x += nv.x; acc.y += nv.y;
                    acc.z += nv.z; acc.w += nv.w;
                }
            }
        }
        int c0 = lane * 4;
        __bf16 hx = (__bf16)acc.x, hy = (__bf16)acc.y;
        __bf16 hz = (__bf16)acc.z, hw = (__bf16)acc.w;
        __bf16* dHi = hHi + m * RS + c0;
        __bf16* dLo = hLo + m * RS + c0;
        dHi[0] = hx; dHi[1] = hy; dHi[2] = hz; dHi[3] = hw;
        dLo[0] = (__bf16)(acc.x - (float)hx);
        dLo[1] = (__bf16)(acc.y - (float)hy);
        dLo[2] = (__bf16)(acc.z - (float)hz);
        dLo[3] = (__bf16)(acc.w - (float)hw);
    }
    __syncthreads();

    // ---------- Phase C: out = h @ W^T + b via bf16x3 WMMA ----------
    const int m0     = wv * 16;
    const int arow   = m0 + (lane & 15);            // A: lane -> M row
    const int kbaseA = (lane & 16) ? 8 : 0;         // A: half-wave K offset
    const int nlocal = lane & 15;                   // C/D: lane -> N column
    const int mAdd   = (lane & 16) ? 8 : 0;         // C/D: half-wave M offset

    for (int nt = 0; nt < 8; ++nt) {
        int n0 = nt * 16;
        float bv = bias[n0 + nlocal];
        v8f acc = { bv, bv, bv, bv, bv, bv, bv, bv };
        #pragma unroll
        for (int kkc = 0; kkc < 4; ++kkc) {
            FragBF aHi, aLo, bHi, bLo;
            // A fragment: two contiguous 16B groups (K = kbase..+7, kbase+16..+23)
            const char* pah = (const char*)(hHi + arow * RS + kkc * 32 + kbaseA);
            aHi.q[0] = *(const uint4*)(pah);
            aHi.q[1] = *(const uint4*)(pah + 32);
            const char* pal = (const char*)(hLo + arow * RS + kkc * 32 + kbaseA);
            aLo.q[0] = *(const uint4*)(pal);
            aLo.q[1] = *(const uint4*)(pal + 32);
            // B fragment: 32B contiguous per lane (pre-packed), conflict-free
            const uint4* pbh = (const uint4*)(wHi + ((nt * 4 + kkc) * 32 + lane) * 16);
            bHi.q[0] = pbh[0]; bHi.q[1] = pbh[1];
            const uint4* pbl = (const uint4*)(wLo + ((nt * 4 + kkc) * 32 + lane) * 16);
            bLo.q[0] = pbl[0]; bLo.q[1] = pbl[1];
            // bf16x3: hi*hi + hi*lo + lo*hi  (~fp32 accuracy)
            acc = __builtin_amdgcn_wmma_f32_16x16x32_bf16(false, aHi.v, false, bHi.v,
                                                          (short)0, acc, false, false);
            acc = __builtin_amdgcn_wmma_f32_16x16x32_bf16(false, aHi.v, false, bLo.v,
                                                          (short)0, acc, false, false);
            acc = __builtin_amdgcn_wmma_f32_16x16x32_bf16(false, aLo.v, false, bHi.v,
                                                          (short)0, acc, false, false);
        }
        #pragma unroll
        for (int r = 0; r < 8; ++r) {
            int g = blockBase + m0 + mAdd + r;
            if (g < NNODES)
                __builtin_nontemporal_store(acc[r], &out[g * COUT + n0 + nlocal]);
        }
    }
}

extern "C" void kernel_launch(void* const* d_in, const int* in_sizes, int n_in,
                              void* d_out, int out_size, void* d_ws, size_t ws_size,
                              hipStream_t stream) {
    const float*     x    = (const float*)d_in[0];
    const long long* eidx = (const long long*)d_in[1];   // int64 per reference
    const float*     eps  = (const float*)d_in[2];
    const float*     W    = (const float*)d_in[3];
    const float*     bias = (const float*)d_in[4];
    float*           out  = (float*)d_out;

    const int grid = (NNODES + TILE_M - 1) / TILE_M;     // 782 blocks
    const size_t smemBytes = (size_t)(2 * TILE_M * RS + 2 * WPACK_ELEMS) * sizeof(__bf16); // 135168 B

    gin_fused_kernel<<<grid, THREADS, smemBytes, stream>>>(x, eidx, eps, W, bias, out);
}
